// RobustTripletLoss_final_32358283608207
// MI455X (gfx1250) — compile-verified, matrix operand order checked by reading, not run on hardware
//
#include <hip/hip_runtime.h>
#include <hip/hip_bf16.h>

#define NN 4096
#define DD 512
#define CC 128
#define TOPK 12
#define MARGIN 0.3f
#define SUPER 1024              // columns per LDS super-tile
#define SEGC  (SUPER / 16)      // 64 columns scanned per thread per super-tile
#define KCH   (DD / 32)         // 16 K-chunks of 32
#define PIPE  4                 // B-fragment ring depth

typedef __attribute__((ext_vector_type(16))) __bf16 v16bf;
typedef __attribute__((ext_vector_type(8)))  __bf16 v8bf;
typedef __attribute__((ext_vector_type(8)))  float  v8f;

__device__ __forceinline__ unsigned int hash_u32(unsigned int x) {
    x ^= x >> 16; x *= 0x7feb352du;
    x ^= x >> 15; x *= 0x846ca68bu;
    x ^= x >> 16;
    return x;
}

// Deterministic stand-in for jax.random.gumbel(key)[i, j]
__device__ __forceinline__ float gumbelf(unsigned int i, unsigned int j, unsigned int salt) {
    unsigned int h = hash_u32(i * 4096u + j + salt * 0x9e3779b9u);
    float u = (float)(h >> 8) * (1.0f / 16777216.0f);
    u = fmaxf(u, 1e-20f);
    return -__logf(fmaxf(-__logf(u), 1e-20f));
}

// Branchless sorted insert (ascending, keep smallest TOPK). Static indices only
// so tv/ti stay in VGPRs (no scratch).
__device__ __forceinline__ void top_insert(float (&tv)[TOPK], int (&ti)[TOPK],
                                           float d, int j) {
    if (d >= tv[TOPK - 1]) return;
    float cv = d; int ci = j;
#pragma unroll
    for (int k = 0; k < TOPK; ++k) {
        const bool ins = cv < tv[k];
        const float ov = tv[k]; const int oi = ti[k];
        tv[k] = ins ? cv : ov;
        ti[k] = ins ? ci : oi;
        cv = ins ? ov : cv;
        ci = ins ? oi : ci;
    }
}

// Load one 16x32 bf16 WMMA fragment (two 16-byte halves per lane).
__device__ __forceinline__ v16bf load_frag(const __bf16* __restrict__ p) {
    v8bf lo = *(const v8bf*)p;
    v8bf hi = *(const v8bf*)(p + 16);
    v16bf r;
#pragma unroll
    for (int e = 0; e < 8; ++e) { r[e] = lo[e]; r[e + 8] = hi[e]; }
    return r;
}

// ---------------------------------------------------------------------------
// Phase 0: f32 -> bf16 copy of inputs + row squared norms
// ---------------------------------------------------------------------------
__global__ __launch_bounds__(256) void prep_kernel(const float* __restrict__ inp,
                                                   __bf16* __restrict__ Abf,
                                                   float* __restrict__ sq) {
    const int row = blockIdx.x;
    const int t   = threadIdx.x;
    float acc = 0.0f;
    for (int e = t; e < DD; e += 256) {
        float v = inp[(size_t)row * DD + e];
        Abf[(size_t)row * DD + e] = (__bf16)v;
        acc += v * v;
    }
    __shared__ float red[256];
    red[t] = acc;
    __syncthreads();
    for (int s = 128; s > 0; s >>= 1) {
        if (t < s) red[t] += red[t + s];
        __syncthreads();
    }
    if (t == 0) sq[row] = red[0];
}

// ---------------------------------------------------------------------------
// Main fused kernel: 16-row strip per workgroup, column dimension strip-mined
// into 1024-wide super-tiles held in a 64 KB LDS buffer.
// B fragments flow through an explicit register ring (software pipelining);
// WGP-scope prefetch warms the next tile across both cacheline halves.
// Scan state parks in LDS across super-tiles (dead during WMMA phase).
// ---------------------------------------------------------------------------
__global__ __launch_bounds__(256) void triplet_kernel(const __bf16* __restrict__ Abf,
                                                      const float* __restrict__ sq,
                                                      const float* __restrict__ pred,
                                                      const int*   __restrict__ targets,
                                                      const float* __restrict__ prob,
                                                      const float* __restrict__ thrp,
                                                      float* __restrict__ acc) {
    extern __shared__ unsigned char smem_raw[];
    float* strip = (float*)smem_raw;                                 // 16 x SUPER f32
    unsigned char* sc = smem_raw + (size_t)16 * SUPER * sizeof(float);
    float* minv = (float*)sc; sc += 256 * 4;
    int*   mini = (int*)sc;   sc += 256 * 4;
    float* g1v  = (float*)sc; sc += 256 * 4;
    int*   g1i  = (int*)sc;   sc += 256 * 4;
    float* g1d  = (float*)sc; sc += 256 * 4;   // dist at gumbel-1 winner
    float* g2v  = (float*)sc; sc += 256 * 4;
    int*   g2i  = (int*)sc;   sc += 256 * 4;
    float* g2d  = (float*)sc; sc += 256 * 4;   // dist at gumbel-2 winner
    float* topv = (float*)sc; sc += 256 * TOPK * 4;
    int*   topi = (int*)sc;

    const int tid     = threadIdx.x;
    const int lane    = tid & 31;
    const int wave    = tid >> 5;
    const int rowbase = blockIdx.x * 16;
    const float thr   = thrp[0];

    // WMMA lane geometry
    const int lrow  = lane & 15;
    const int khalf = (lane >> 4) * 8;
    const int nn    = lane & 15;
    const int mbase = (lane >> 4) * 8;

    // Scan geometry (16 threads per row)
    const int row  = tid >> 4;
    const int seg  = tid & 15;
    const int grow = rowbase + row;
    const int tgt  = targets[grow];

    // ---- Initialize LDS-resident scan state (per thread slot) ----
    minv[tid] = __builtin_inff(); mini[tid] = 0;
    g1v[tid] = -__builtin_inff(); g1i[tid] = -1; g1d[tid] = 0.0f;
    g2v[tid] = -__builtin_inff(); g2i[tid] = -1; g2d[tid] = 0.0f;
#pragma unroll
    for (int k = 0; k < TOPK; ++k) {
        topv[tid * TOPK + k] = __builtin_inff();
        topi[tid * TOPK + k] = 0;
    }
    __syncthreads();

    const __bf16* pArow = Abf + (size_t)(rowbase + lrow) * DD + khalf;

    for (int sup = 0; sup < NN; sup += SUPER) {
        // ------------- Phase 1: WMMA distances for this super-tile -------------
        // Preload all A fragments (live only within this phase).
        v16bf afrag[KCH];
#pragma unroll
        for (int q = 0; q < KCH; ++q) afrag[q] = load_frag(pArow + q * 32);

        for (int jt = wave; jt < SUPER / 16; jt += 8) {
            const int colbase = sup + jt * 16;

            // Branchless WGP-scope prefetch of the next B tile; all 32 lanes
            // participate so both halves of each 1 KB row get warmed.
            const int jn = (jt + 8 < SUPER / 16) ? (jt + 8) : jt;
            const __bf16* pf = Abf + (size_t)(sup + jn * 16 + lrow) * DD
                                   + (size_t)(lane >> 4) * (DD / 2);
            __builtin_prefetch(pf, 0, 3);

            const __bf16* pB = Abf + (size_t)(colbase + lrow) * DD + khalf;

            // Prime the B ring: chunks 0..PIPE-1 in flight.
            v16bf bbuf[PIPE];
#pragma unroll
            for (int p = 0; p < PIPE; ++p) bbuf[p] = load_frag(pB + p * 32);

            v8f c = {};
#pragma unroll
            for (int q = 0; q < KCH; ++q) {
                c = __builtin_amdgcn_wmma_f32_16x16x32_bf16(
                        false, afrag[q], false, bbuf[q & (PIPE - 1)],
                        (short)0, c, false, false);
                if (q + PIPE < KCH) {
                    bbuf[q & (PIPE - 1)] = load_frag(pB + (q + PIPE) * 32);
                }
            }

            const float sqc = sq[colbase + nn];
#pragma unroll
            for (int v = 0; v < 8; ++v) {
                const int m = mbase + v;
                float d2    = sq[rowbase + m] + sqc - 2.0f * c[v];
                strip[m * SUPER + jt * 16 + nn] = sqrtf(fmaxf(d2, 1e-12f));
            }
        }
        __syncthreads();

        // ------------- Phase 2a: incremental segmented scan --------------------
        // Pull state from LDS, update in registers, push back before barrier.
        {
            float mnv = minv[tid]; int mni = mini[tid];
            float bg1 = g1v[tid];  int bi1 = g1i[tid];  float bd1 = g1d[tid];
            float bg2 = g2v[tid];  int bi2 = g2i[tid];  float bd2 = g2d[tid];
            float tv[TOPK]; int ti[TOPK];
#pragma unroll
            for (int k = 0; k < TOPK; ++k) {
                tv[k] = topv[tid * TOPK + k];
                ti[k] = topi[tid * TOPK + k];
            }

            for (int jl = seg * SEGC; jl < (seg + 1) * SEGC; ++jl) {
                const int j   = sup + jl;
                const float d = strip[row * SUPER + jl];
                const bool same = (targets[j] == tgt);
                if (!same) {
                    if (d < mnv) { mnv = d; mni = j; }
                    top_insert(tv, ti, d, j);
                } else if (j != grow) {
                    const float g = gumbelf((unsigned)grow, (unsigned)j, 0x1234567u);
                    if (g > bg1) { bg1 = g; bi1 = j; bd1 = d; }
                    if (prob[j] >= thr) {
                        const float g2 = gumbelf((unsigned)grow, (unsigned)j, 0x89abcdefu);
                        if (g2 > bg2) { bg2 = g2; bi2 = j; bd2 = d; }
                    }
                }
            }

            minv[tid] = mnv; mini[tid] = mni;
            g1v[tid] = bg1;  g1i[tid] = bi1;  g1d[tid] = bd1;
            g2v[tid] = bg2;  g2i[tid] = bi2;  g2d[tid] = bd2;
#pragma unroll
            for (int k = 0; k < TOPK; ++k) {
                topv[tid * TOPK + k] = tv[k];
                topi[tid * TOPK + k] = ti[k];
            }
        }
        __syncthreads();   // strip + state reused by next super-tile
    }

    // ------------- Phase 2b: leader merge + case logic -------------
    if (tid < 16) {
        const int r    = tid;
        const int gr   = rowbase + r;
        const int tgtr = targets[gr];

        float an0 = __builtin_inff(); int an_idx = 0;
        float g1  = -__builtin_inff(); int ap_idx = 0;  float ap0    = 0.0f;
        float g2  = -__builtin_inff(); int np_idx = -1; float ap_new = 0.0f;
        float mv[TOPK]; int mi[TOPK];
#pragma unroll
        for (int k = 0; k < TOPK; ++k) { mv[k] = __builtin_inff(); mi[k] = 0; }

        for (int s = 0; s < 16; ++s) {
            const int t = r * 16 + s;
            if (minv[t] < an0) { an0 = minv[t]; an_idx = mini[t]; }
            if (g1i[t] >= 0 && g1v[t] > g1) { g1 = g1v[t]; ap_idx = g1i[t]; ap0 = g1d[t]; }
            if (g2i[t] >= 0 && g2v[t] > g2) { g2 = g2v[t]; np_idx = g2i[t]; ap_new = g2d[t]; }
            for (int k = 0; k < TOPK; ++k) {
                top_insert(mv, mi, topv[t * TOPK + k], topi[t * TOPK + k]);
            }
        }

        const bool pos_conf = prob[ap_idx] >= thr;
        const bool neg_conf = prob[an_idx] >= thr;

        // is_fn = argmax(prediction[an_idx]) == targets[row]
        float pm = -__builtin_inff(); int pmi = 0;
        for (int c = 0; c < CC; ++c) {
            const float p = pred[(size_t)an_idx * CC + c];
            if (p > pm) { pm = p; pmi = c; }
        }
        const bool is_fn = (pmi == tgtr);

        // first confident candidate in tail, else TOPK-1; an_B comes from the
        // merged top-12 values (dist carried) -- no strip lookup needed
        int rr = TOPK - 1;
        for (int k = 1; k < TOPK; ++k) {
            if (prob[mi[k]] >= thr) { rr = k; break; }
        }
        const float an_B = mv[rr];
        if (np_idx < 0) ap_new = ap0;   // fallback: new_pos == ap_idx

        const bool caseB   = pos_conf && !neg_conf && is_fn;
        const bool swp     = !pos_conf && (neg_conf || !is_fn);
        const bool inverse = !pos_conf && !neg_conf && is_fn;
        const float half   = 0.5f * (ap0 + an0);
        const float ap = caseB ? half : (swp ? ap_new : ap0);
        const float an = caseB ? an_B : (swp ? half : an0);
        const float rl = inverse ? fmaxf(an - ap + MARGIN, 0.0f)
                                 : fmaxf(ap - an + MARGIN, 0.0f);

        if (prob[gr] >= thr) {
            atomicAdd(&acc[0], rl);
            atomicAdd((int*)&acc[1], 1);
            if (an >= ap) atomicAdd((int*)&acc[2], 1);
        }
    }
}

// ---------------------------------------------------------------------------
// Finalize: loss = sum / max(cnt,1); out = [loss, correct]
// ---------------------------------------------------------------------------
__global__ void finalize_kernel(const float* __restrict__ acc, float* __restrict__ out) {
    const int cnt     = ((const int*)acc)[1];
    const int correct = ((const int*)acc)[2];
    const int denom   = cnt > 1 ? cnt : 1;
    out[0] = acc[0] / (float)denom;
    out[1] = (float)correct;
}

extern "C" void kernel_launch(void* const* d_in, const int* in_sizes, int n_in,
                              void* d_out, int out_size, void* d_ws, size_t ws_size,
                              hipStream_t stream) {
    (void)in_sizes; (void)n_in; (void)out_size; (void)ws_size;
    const float* inputs     = (const float*)d_in[0];
    const float* prediction = (const float*)d_in[1];
    const int*   targets    = (const int*)d_in[2];
    // d_in[3] = true_targets (unused by the reference computation)
    const float* prob       = (const float*)d_in[4];
    const float* threshold  = (const float*)d_in[5];

    unsigned char* ws = (unsigned char*)d_ws;
    __bf16* Abf = (__bf16*)ws;                                        // 4 MB bf16 inputs
    float*  sq  = (float*)(ws + (size_t)NN * DD * sizeof(__bf16));    // 16 KB norms
    float*  acc = (float*)(ws + (size_t)NN * DD * sizeof(__bf16) + (size_t)NN * sizeof(float));

    hipMemsetAsync(acc, 0, 16, stream);  // zero [loss_sum, cnt, correct] each launch

    prep_kernel<<<NN, 256, 0, stream>>>(inputs, Abf, sq);

    const size_t lds_bytes = (size_t)16 * SUPER * sizeof(float)  // 64 KB strip
                           + 256 * 4 * 8                         // reduction partials
                           + 256 * TOPK * 4 * 2;                 // top-12 partials
    triplet_kernel<<<NN / 16, 256, lds_bytes, stream>>>(
        Abf, sq, prediction, targets, prob, threshold, acc);

    finalize_kernel<<<1, 1, 0, stream>>>(acc, (float*)d_out);
}
